// ColorGNNSmallEmbedding_59287728554039
// MI455X (gfx1250) — matrix operand order repaired and checked by reading, not compiled, not run on hardware
//
#include <hip/hip_runtime.h>
#include <cstdint>
#include <cstddef>

// MI455X (gfx1250) implementation.
// Roofline: dominant GEMM resnet(100000x1000)@W1slice = 25.6 GFLOP, memory
// floor = read x once (402 MB @ 23.3 TB/s ~= 17us) -> bf16 WMMA w/ f32 acc.
// Graph aggregation done gather-style via device-built CSR to avoid ~205M
// f32 atomics (which would dominate at L2 atomic throughput).

typedef __attribute__((ext_vector_type(16))) __bf16 v16bf;
typedef __attribute__((ext_vector_type(8)))  float  v8f;

#define NODE_DIM 1005
#define RESNET_K 1000
#define H1 128
#define H2 32

static __device__ __forceinline__ __bf16 f2bf(float f) { return (__bf16)f; }

// ---------------------------------------------------------------------------
// Prep: pack W1[250:1250,0:128] and W2 into WMMA-B fragment order (bf16).
// B frag (16-bit, 32x16): lane holds N = nt*16 + (lane&15),
// K = kstep*32 + (lane>>4)*8 + (h<8 ? h : h+8)  for half h in [0,16).
// Flat layout: [((kstep*NT + nt)*32 + lane)*16 + h]  -> 32 contiguous bytes (2x b128) per lane.
// ---------------------------------------------------------------------------
__global__ __launch_bounds__(256) void prep_wfrag_kernel(
    const float* __restrict__ W1, const float* __restrict__ W2,
    __bf16* __restrict__ w1f, __bf16* __restrict__ w2f)
{
  int tid = blockIdx.x * blockDim.x + threadIdx.x;
  if (tid < 32 * 8 * 32) {                      // W1: 32 ksteps x 8 ntiles
    int lane = tid & 31;
    int nt   = (tid >> 5) & 7;
    int ks   = tid >> 8;
    int col  = nt * 16 + (lane & 15);
    int kb   = ks * 32 + ((lane >> 4) * 8);
    #pragma unroll
    for (int h = 0; h < 16; ++h) {
      int k = kb + (h < 8 ? h : h + 8);
      float v = (k < RESNET_K) ? W1[(size_t)(250 + k) * H1 + col] : 0.0f;
      w1f[(size_t)tid * 16 + h] = f2bf(v);
    }
  } else if (tid < 32 * 8 * 32 + 4 * 2 * 32) {  // W2: 4 ksteps x 2 ntiles
    int t2   = tid - 32 * 8 * 32;
    int lane = t2 & 31;
    int nt   = (t2 >> 5) & 1;
    int ks   = t2 >> 6;
    int col  = nt * 16 + (lane & 15);
    int kb   = ks * 32 + ((lane >> 4) * 8);
    #pragma unroll
    for (int h = 0; h < 16; ++h) {
      int k = kb + (h < 8 ? h : h + 8);
      w2f[(size_t)t2 * 16 + h] = f2bf(W2[(size_t)k * H2 + col]);
    }
  }
}

// ---------------------------------------------------------------------------
// Prep: fold embedding tables through W1 -> small lookup tables (fp32, exact).
// rows: 3 (layer) + 11 (rel) + 3*256 (color slots) = 782, each 128 cols.
// ---------------------------------------------------------------------------
__global__ __launch_bounds__(128) void prep_tables_kernel(
    const float* __restrict__ embL, const float* __restrict__ embR,
    const float* __restrict__ embC, const float* __restrict__ W1,
    float* __restrict__ tL, float* __restrict__ tR, float* __restrict__ tC)
{
  int tid = blockIdx.x * blockDim.x + threadIdx.x;
  if (tid >= 782 * 128) return;
  int r = tid >> 7, c = tid & 127;
  if (r < 3) {
    float s = 0.f;
    for (int j = 0; j < 250; ++j) s += embL[r * 250 + j] * W1[(size_t)j * H1 + c];
    tL[r * 128 + c] = s;
  } else if (r < 14) {
    int rr = r - 3;
    float s = 0.f;
    for (int j = 0; j < 250; ++j) s += embR[rr * 250 + j] * W1[(size_t)(1250 + j) * H1 + c];
    tR[rr * 128 + c] = s;
  } else {
    int rr = r - 14;            // rr = kpos*256 + value
    int kpos = rr >> 8, v = rr & 255;
    float s = 0.f;
    for (int j = 0; j < 85; ++j)
      s += embC[v * 85 + j] * W1[(size_t)(1500 + kpos * 85 + j) * H1 + c];
    tC[(size_t)rr * 128 + c] = s;
  }
}

// ---------------------------------------------------------------------------
// Prefill xw1 with the embedding-table contribution (C-init for GEMM1).
// ---------------------------------------------------------------------------
__global__ __launch_bounds__(256) void emb_prefill_kernel(
    const float* __restrict__ x, const float* __restrict__ tL,
    const float* __restrict__ tR, const float* __restrict__ tC,
    float* __restrict__ xw1, int n_nodes)
{
  int tid = blockIdx.x * blockDim.x + threadIdx.x;
  int node = tid >> 7;
  if (node >= n_nodes) return;
  int c = tid & 127;
  const float* xr = x + (size_t)node * NODE_DIM;
  int lid = (int)xr[0];
  lid = lid < 0 ? 0 : (lid > 2 ? 2 : lid);
  int rid = (int)roundf(xr[1001] * 10.0f);
  rid = rid < 0 ? 0 : (rid > 10 ? 10 : rid);
  int c0 = (int)xr[1002]; c0 = c0 < 0 ? 0 : (c0 > 255 ? 255 : c0);
  int c1 = (int)xr[1003]; c1 = c1 < 0 ? 0 : (c1 > 255 ? 255 : c1);
  int c2 = (int)xr[1004]; c2 = c2 < 0 ? 0 : (c2 > 255 ? 255 : c2);
  xw1[(size_t)node * H1 + c] =
      tL[lid * 128 + c] + tR[rid * 128 + c] +
      tC[(size_t)(0 * 256 + c0) * 128 + c] +
      tC[(size_t)(1 * 256 + c1) * 128 + c] +
      tC[(size_t)(2 * 256 + c2) * 128 + c];
}

// ---------------------------------------------------------------------------
// Graph prep: weighted degree (self loop = +1) + CSR-by-dst build.
// ---------------------------------------------------------------------------
__global__ __launch_bounds__(256) void graph_init_kernel(
    float* __restrict__ deg, int* __restrict__ cnt, int n_nodes)
{
  int i = blockIdx.x * blockDim.x + threadIdx.x;
  if (i < n_nodes) { deg[i] = 1.0f; cnt[i] = 0; }
}

__global__ __launch_bounds__(256) void deg_cnt_kernel(
    const int* __restrict__ dst, const float* __restrict__ ew,
    float* __restrict__ deg, int* __restrict__ cnt, int n_edges)
{
  int e = blockIdx.x * blockDim.x + threadIdx.x;
  if (e >= n_edges) return;
  int d = dst[e];
  atomicAdd(&deg[d], ew[e]);
  atomicAdd(&cnt[d], 1);
}

__global__ __launch_bounds__(1024) void scan_kernel(
    const int* __restrict__ cnt, int* __restrict__ rowptr,
    int* __restrict__ pos, int n)
{
  __shared__ int part[1024];
  int t = threadIdx.x;
  int chunk = (n + 1023) / 1024;
  int lo = t * chunk;
  int hi = lo + chunk; if (hi > n) hi = n;
  int s = 0;
  for (int i = lo; i < hi; ++i) s += cnt[i];
  part[t] = s;
  __syncthreads();
  for (int off = 1; off < 1024; off <<= 1) {
    int v = (t >= off) ? part[t - off] : 0;
    __syncthreads();
    part[t] += v;
    __syncthreads();
  }
  int run = (t == 0) ? 0 : part[t - 1];
  for (int i = lo; i < hi; ++i) {
    rowptr[i] = run; pos[i] = run;
    run += cnt[i];
  }
  if (t == 1023) rowptr[n] = run;
}

__global__ __launch_bounds__(256) void bucket_fill_kernel(
    const int* __restrict__ dst, int* __restrict__ pos,
    int* __restrict__ eidx, int n_edges)
{
  int e = blockIdx.x * blockDim.x + threadIdx.x;
  if (e >= n_edges) return;
  int slot = atomicAdd(&pos[dst[e]], 1);
  eidx[slot] = e;
}

__global__ __launch_bounds__(256) void rsqrt_kernel(float* __restrict__ deg, int n)
{
  int i = blockIdx.x * blockDim.x + threadIdx.x;
  if (i < n) deg[i] = rsqrtf(deg[i]);   // deg >= 1 always (self loop)
}

// ---------------------------------------------------------------------------
// GEMM1: xw1 += resnet(100000x1000) @ W1[250:1250] via bf16 WMMA 16x16x32.
// One wave computes a 16x128 tile (8 N-tiles, 64 acc VGPRs). 8 waves/block.
// A frag (16-bit A 16x32): lane row = lane&15, K runs [kb0,kb0+8) and
// [kb0+16,kb0+24), kb0 = (lane>>4)*8; fp32 loads (dword, x row stride 1005),
// nontemporal (x is streamed once), converted to bf16.
// ---------------------------------------------------------------------------
__global__ __launch_bounds__(256) void gemm1_kernel(
    const float* __restrict__ x, const __bf16* __restrict__ w1f,
    float* __restrict__ xw1, int n_nodes)
{
  const int lane = threadIdx.x & 31;
  const int wave = threadIdx.x >> 5;
  const int rowbase = blockIdx.x * 128 + wave * 16;
  int arow = rowbase + (lane & 15);
  if (arow >= n_nodes) arow = n_nodes - 1;
  const float* xrow = x + (size_t)arow * NODE_DIM + 1;   // resnet features
  const int kb0 = (lane >> 4) * 8;

  v8f zero = {};
  v8f acc[8];
  #pragma unroll
  for (int t = 0; t < 8; ++t) acc[t] = zero;

  for (int ks = 0; ks < 32; ++ks) {
    const int s0 = ks * 32 + kb0;
    const int s1 = s0 + 16;
    const bool ok0 = (s0 < RESNET_K);   // runs are fully valid or fully OOB
    const bool ok1 = (s1 < RESNET_K);
    v16bf a;
    #pragma unroll
    for (int i = 0; i < 8; ++i) {
      float f0 = ok0 ? __builtin_nontemporal_load(xrow + s0 + i) : 0.0f;
      float f1 = ok1 ? __builtin_nontemporal_load(xrow + s1 + i) : 0.0f;
      a[i]     = f2bf(f0);
      a[8 + i] = f2bf(f1);
    }
    const __bf16* bp = w1f + ((size_t)(ks * 8) * 32 + lane) * 16;
    #pragma unroll
    for (int t = 0; t < 8; ++t) {
      v16bf b = *(const v16bf*)(bp + (size_t)t * 32 * 16);
      acc[t] = __builtin_amdgcn_wmma_f32_16x16x32_bf16(
          false, a, false, b, (short)0, acc[t], false, false);
    }
  }

  // C layout: VGPR r -> M = r + 8*(lane>>4), N = nt*16 + (lane&15).
  const int hi8 = (lane >> 4) * 8;
  const int col = lane & 15;
  #pragma unroll
  for (int r = 0; r < 8; ++r) {
    int row = rowbase + hi8 + r;
    if (row < n_nodes) {
      #pragma unroll
      for (int t = 0; t < 8; ++t) {
        size_t idx = (size_t)row * H1 + t * 16 + col;
        xw1[idx] = xw1[idx] + acc[t][r];   // add embedding prefill
      }
    }
  }
}

// ---------------------------------------------------------------------------
// CSR gather aggregation (symmetric norm, self loop): one wave per node.
// agg[d] = dinv[d]^2 * xw[d] + sum_e dinv[src]*ew*dinv[d] * xw[src]
// ---------------------------------------------------------------------------
template <int COLS>
__global__ __launch_bounds__(256) void agg_gather_kernel(
    const int* __restrict__ rowptr, const int* __restrict__ eidx,
    const int* __restrict__ src, const float* __restrict__ ew,
    const float* __restrict__ dinv, const float* __restrict__ xw,
    float* __restrict__ agg, int n_nodes)
{
  int node = (blockIdx.x * blockDim.x + threadIdx.x) >> 5;
  if (node >= n_nodes) return;
  int lane = threadIdx.x & 31;
  constexpr int V = COLS / 32;
  float di = dinv[node];
  const float* xn = xw + (size_t)node * COLS;
  float acc[V];
  #pragma unroll
  for (int i = 0; i < V; ++i) acc[i] = di * di * xn[lane + 32 * i];
  int lo = rowptr[node], hi = rowptr[node + 1];
  for (int p = lo; p < hi; ++p) {
    int e = eidx[p];
    int s = src[e];
    float coef = dinv[s] * ew[e] * di;
    const float* xs = xw + (size_t)s * COLS;
    #pragma unroll
    for (int i = 0; i < V; ++i) acc[i] += coef * xs[lane + 32 * i];
  }
  float* o = agg + (size_t)node * COLS;
  #pragma unroll
  for (int i = 0; i < V; ++i) o[lane + 32 * i] = acc[i];
}

// ---------------------------------------------------------------------------
// GEMM2: xw2 = relu(agg1 + b1) @ W2(128x32) via bf16 WMMA (bias+relu fused
// into A-fragment build; h1 never materialized). Wave tile = 16x32.
// ---------------------------------------------------------------------------
__global__ __launch_bounds__(256) void gemm2_kernel(
    const float* __restrict__ agg1, const float* __restrict__ b1,
    const __bf16* __restrict__ w2f, float* __restrict__ xw2, int n_nodes)
{
  const int lane = threadIdx.x & 31;
  const int wave = threadIdx.x >> 5;
  const int rowbase = blockIdx.x * 128 + wave * 16;
  int arow = rowbase + (lane & 15);
  if (arow >= n_nodes) arow = n_nodes - 1;
  const float* hrow = agg1 + (size_t)arow * H1;
  const int kb0 = (lane >> 4) * 8;

  v8f zero = {};
  v8f acc[2] = {zero, zero};

  #pragma unroll
  for (int ks = 0; ks < 4; ++ks) {
    const int s0 = ks * 32 + kb0;
    const int s1 = s0 + 16;
    v16bf a;
    #pragma unroll
    for (int i = 0; i < 8; ++i) {
      float f0 = fmaxf(hrow[s0 + i] + b1[s0 + i], 0.0f);
      float f1 = fmaxf(hrow[s1 + i] + b1[s1 + i], 0.0f);
      a[i]     = f2bf(f0);
      a[8 + i] = f2bf(f1);
    }
    const __bf16* bp = w2f + ((size_t)(ks * 2) * 32 + lane) * 16;
    #pragma unroll
    for (int t = 0; t < 2; ++t) {
      v16bf b = *(const v16bf*)(bp + (size_t)t * 32 * 16);
      acc[t] = __builtin_amdgcn_wmma_f32_16x16x32_bf16(
          false, a, false, b, (short)0, acc[t], false, false);
    }
  }

  const int hi8 = (lane >> 4) * 8;
  const int col = lane & 15;
  #pragma unroll
  for (int r = 0; r < 8; ++r) {
    int row = rowbase + hi8 + r;
    if (row < n_nodes) {
      #pragma unroll
      for (int t = 0; t < 2; ++t)
        xw2[(size_t)row * H2 + t * 16 + col] = acc[t][r];
    }
  }
}

// ---------------------------------------------------------------------------
// Final: out = relu(agg2 + b2) @ Wp(32x3) + bp   (tiny, scalar VALU)
// ---------------------------------------------------------------------------
__global__ __launch_bounds__(256) void final_proj_kernel(
    const float* __restrict__ agg2, const float* __restrict__ b2,
    const float* __restrict__ Wp, const float* __restrict__ bp,
    float* __restrict__ out, int n_nodes)
{
  int node = blockIdx.x * blockDim.x + threadIdx.x;
  if (node >= n_nodes) return;
  const float* a = agg2 + (size_t)node * H2;
  float o0 = bp[0], o1 = bp[1], o2 = bp[2];
  #pragma unroll
  for (int c = 0; c < H2; ++c) {
    float h = fmaxf(a[c] + b2[c], 0.0f);
    o0 += h * Wp[c * 3 + 0];
    o1 += h * Wp[c * 3 + 1];
    o2 += h * Wp[c * 3 + 2];
  }
  out[(size_t)node * 3 + 0] = o0;
  out[(size_t)node * 3 + 1] = o1;
  out[(size_t)node * 3 + 2] = o2;
}

// ---------------------------------------------------------------------------
extern "C" void kernel_launch(void* const* d_in, const int* in_sizes, int n_in,
                              void* d_out, int out_size, void* d_ws, size_t ws_size,
                              hipStream_t stream)
{
  const float* x    = (const float*)d_in[0];
  const int*   ei   = (const int*)d_in[1];
  const float* ew   = (const float*)d_in[2];
  const float* embL = (const float*)d_in[3];
  const float* embR = (const float*)d_in[4];
  const float* embC = (const float*)d_in[5];
  const float* W1   = (const float*)d_in[6];
  const float* b1   = (const float*)d_in[7];
  const float* W2   = (const float*)d_in[8];
  const float* b2   = (const float*)d_in[9];
  const float* Wp   = (const float*)d_in[10];
  const float* bp   = (const float*)d_in[11];
  float* out = (float*)d_out;

  const int n_nodes = in_sizes[0] / NODE_DIM;
  const int n_edges = in_sizes[1] / 2;
  const int* src = ei;
  const int* dst = ei + n_edges;

  // workspace bump allocator (256B aligned)
  char* p = (char*)d_ws;
  size_t used = 0;
  auto alloc = [&](size_t bytes) -> void* {
    void* r = p + used;
    used += (bytes + 255) & ~(size_t)255;
    return r;
  };
  __bf16* w1f   = (__bf16*)alloc((size_t)32 * 8 * 32 * 16 * 2);
  __bf16* w2f   = (__bf16*)alloc((size_t)4 * 2 * 32 * 16 * 2);
  float*  tL    = (float*)alloc((size_t)3 * 128 * 4);
  float*  tR    = (float*)alloc((size_t)11 * 128 * 4);
  float*  tC    = (float*)alloc((size_t)3 * 256 * 128 * 4);
  float*  dinv  = (float*)alloc((size_t)n_nodes * 4);       // deg, then rsqrt
  int*    cnt   = (int*)alloc((size_t)n_nodes * 4);
  int*    rowp  = (int*)alloc((size_t)(n_nodes + 1) * 4);
  int*    pos   = (int*)alloc((size_t)n_nodes * 4);
  int*    eidx  = (int*)alloc((size_t)n_edges * 4);
  float*  xw1   = (float*)alloc((size_t)n_nodes * H1 * 4);
  float*  agg1  = (float*)alloc((size_t)n_nodes * H1 * 4);
  float*  xw2   = (float*)alloc((size_t)n_nodes * H2 * 4);
  float*  agg2  = (float*)alloc((size_t)n_nodes * H2 * 4);
  if (used > ws_size) return;   // workspace too small: bail safely

  const int TB = 256;
  const int eb   = (n_edges + TB - 1) / TB;
  const int nb   = (n_nodes + TB - 1) / TB;
  const int gemb = (n_nodes + 127) / 128;                   // 128 rows / block

  // 1) weight fragment packing + embedding-table folding
  prep_wfrag_kernel<<<(32*8*32 + 4*2*32 + TB - 1) / TB, TB, 0, stream>>>(W1, W2, w1f, w2f);
  prep_tables_kernel<<<782, 128, 0, stream>>>(embL, embR, embC, W1, tL, tR, tC);

  // 2) embedding prefill (C-init of GEMM1)
  emb_prefill_kernel<<<(n_nodes * 128 + TB - 1) / TB, TB, 0, stream>>>(
      x, tL, tR, tC, xw1, n_nodes);

  // 3) degree + CSR build
  graph_init_kernel<<<nb, TB, 0, stream>>>(dinv, cnt, n_nodes);
  deg_cnt_kernel<<<eb, TB, 0, stream>>>(dst, ew, dinv, cnt, n_edges);
  scan_kernel<<<1, 1024, 0, stream>>>(cnt, rowp, pos, n_nodes);
  bucket_fill_kernel<<<eb, TB, 0, stream>>>(dst, pos, eidx, n_edges);
  rsqrt_kernel<<<nb, TB, 0, stream>>>(dinv, n_nodes);

  // 4) layer 1: WMMA GEMM then CSR gather aggregation
  gemm1_kernel<<<gemb, TB, 0, stream>>>(x, w1f, xw1, n_nodes);
  agg_gather_kernel<128><<<(n_nodes * 32 + TB - 1) / TB, TB, 0, stream>>>(
      rowp, eidx, src, ew, dinv, xw1, agg1, n_nodes);

  // 5) layer 2: fused bias+relu WMMA GEMM then aggregation
  gemm2_kernel<<<gemb, TB, 0, stream>>>(agg1, b1, w2f, xw2, n_nodes);
  agg_gather_kernel<32><<<(n_nodes * 32 + TB - 1) / TB, TB, 0, stream>>>(
      rowp, eidx, src, ew, dinv, xw2, agg2, n_nodes);

  // 6) final projection
  final_proj_kernel<<<nb, TB, 0, stream>>>(agg2, b2, Wp, bp, out, n_nodes);
}
